// KMeansPalettizedConv2d_59889023975690
// MI455X (gfx1250) — matrix operand order repaired and checked by reading, not compiled, not run on hardware
//
#include <hip/hip_runtime.h>
#include <hip/hip_bf16.h>
#include <stdint.h>

typedef __attribute__((ext_vector_type(16))) __bf16 v16bf;
typedef __attribute__((ext_vector_type(8)))  float  v8f;

#define O_CH   256
#define C_CH   256
#define HWDIM  56
#define HWSQ   (56 * 56)              // 3136
#define PADW   58
#define PADSQ  (58 * 58)              // 3364
#define KDIM   (C_CH * 9)             // 2304
#define KSTEP  32
#define KITERS (KDIM / KSTEP)         // 72
#define NTILE  128                    // spatial tile (M tile = all 256 out channels)
#define PITCH  40                     // u16 per LDS row: 32 data + 8 pad (80 B, 16B-aligned)
#define IN_MAX (32 * C_CH * HWSQ - 1) // clamp bound for fallback speculative loads

// fp32 -> bf16 bits, round-to-nearest-even
__device__ __forceinline__ uint16_t f2bf(float f) {
    uint32_t u = __float_as_uint(f);
    uint32_t r = u + 0x7FFFu + ((u >> 16) & 1u);
    return (uint16_t)(r >> 16);
}

// Kernel 1: palette -> bf16 weights, PRE-SWIZZLED into WMMA A-fragment order:
//   wswz[((M16*KITERS + kt)*32 + lane)*16 + j]
// holds, for lane 'lane' of fragment (m-tile M16, k-step kt), its 16 bf16:
//   m = M16*16 + (lane&15), hi = lane>>4,
//   j<8 : k = kt*32 + hi*8 + j      (VGPR0-3)
//   j>=8: k = kt*32 + 16 + hi*8 + j-8 (VGPR4-7)
// => each A-fragment load in the GEMM is one fully-coalesced contiguous block.
__global__ void dequant_kernel(const int* __restrict__ widx,
                               const float* __restrict__ lut,
                               uint16_t* __restrict__ wswz, int n) {
    int i = blockIdx.x * blockDim.x + threadIdx.x;
    if (i >= n) return;
    int j   = i & 15;
    int L   = (i >> 4) & 31;
    int t2  = i >> 9;
    int kt  = t2 % KITERS;
    int M16 = t2 / KITERS;
    int m   = M16 * 16 + (L & 15);
    int k   = kt * 32 + (L >> 4) * 8 + (j & 7) + (j >> 3) * 16;
    wswz[i] = f2bf(lut[widx[m * KDIM + k]]);
}

// Kernel 1b: zero-padded bf16 input [32][256][58][58] (one-shot, pure bandwidth).
__global__ void pad_kernel(const float* __restrict__ in,
                           uint16_t* __restrict__ pin, int total) {
    int i = blockIdx.x * blockDim.x + threadIdx.x;
    if (i >= total) return;
    int pl  = i / PADSQ;              // nn*256 + c plane
    int rem = i - pl * PADSQ;
    int y   = rem / PADW;
    int x   = rem - y * PADW;
    float v = 0.f;
    if ((unsigned)(y - 1) < HWDIM && (unsigned)(x - 1) < HWDIM)
        v = in[pl * HWSQ + (y - 1) * HWDIM + (x - 1)];
    pin[i] = f2bf(v);
}

// Kernel 2: implicit-GEMM conv via V_WMMA_F32_16X16X32_BF16.
// Block tile: M=256 x N=128. 8 waves = 4(M) x 2(N); wave tile 64x64 = 4x4 frags.
// A fragments: contiguous loads from the swizzled, L2-resident weight matrix.
// B tile: double-buffered LDS; next tile gathered while current tile is consumed.
template <bool PADDED>
__global__ __launch_bounds__(256)
void conv_wmma_kernel(const float* __restrict__ in,      // !PADDED path
                      const uint16_t* __restrict__ pin,  // PADDED path
                      const uint16_t* __restrict__ wswz,
                      const float* __restrict__ bias,
                      float* __restrict__ out) {
    __shared__ __align__(16) uint16_t Bs[2][NTILE * PITCH];

    const int tid   = threadIdx.x;
    const int lane  = tid & 31;
    const int wave  = tid >> 5;       // 0..7
    const int mWave = wave & 3;       // 4 waves span M=256: 64 rows each
    const int nWave = wave >> 2;      // 2 waves span N=128: 64 cols each
    const int ml    = lane & 15;
    const int hi    = lane >> 4;

    const int nBase = blockIdx.x * NTILE;

    // ---- B-gather role: wave owns k-rows [4*wave, 4*wave+4); lanes sweep n ----
    int baseG[4], hG[4], wG[4];
    #pragma unroll
    for (int g = 0; g < 4; ++g) {
        int p  = nBase + g * 32 + lane;
        int nn = p / HWSQ;
        int hw = p - nn * HWSQ;
        int h  = hw / HWDIM;
        int w  = hw - h * HWDIM;
        if constexpr (PADDED) {
            baseG[g] = nn * (C_CH * PADSQ) + h * PADW + w;
        } else {
            baseG[g] = nn * (C_CH * HWSQ) + h * HWDIM + w;
            hG[g] = h;  wG[g] = w;
        }
    }

    // gather one 128x32 B tile for k-step ktG into LDS buffer b
    auto gather = [&](int ktG, int b) {
        const int kb = ktG * KSTEP + wave * 4;       // wave-uniform k base
        int kOff[4], rm1[4], sm1[4];
        #pragma unroll
        for (int kk = 0; kk < 4; ++kk) {
            int k  = kb + kk;
            int c  = k / 9;
            int rs = k - c * 9;
            int r  = (rs * 11) >> 5;                 // rs/3 for rs in [0,8]
            int s  = rs - r * 3;
            if constexpr (PADDED) {
                kOff[kk] = c * PADSQ + r * PADW + s; // pad absorbs the -1s
            } else {
                rm1[kk] = r - 1;
                sm1[kk] = s - 1;
                kOff[kk] = c * HWSQ + rm1[kk] * HWDIM + sm1[kk];
            }
        }
        #pragma unroll
        for (int g = 0; g < 4; ++g) {
            union { uint16_t u[4]; uint2 q; } pk;
            #pragma unroll
            for (int kk = 0; kk < 4; ++kk) {
                if constexpr (PADDED) {
                    pk.u[kk] = pin[baseG[g] + kOff[kk]];   // 1 add + u16 load
                } else {
                    int ih  = hG[g] + rm1[kk];
                    int iw  = wG[g] + sm1[kk];
                    bool ok = ((unsigned)ih < HWDIM) & ((unsigned)iw < HWDIM);
                    int off = baseG[g] + kOff[kk];
                    off = off < 0 ? 0 : (off > IN_MAX ? IN_MAX : off);
                    float v = in[off];
                    v = ok ? v : 0.f;
                    pk.u[kk] = __builtin_bit_cast(uint16_t, (__bf16)v);
                }
            }
            *(uint2*)&Bs[b][(g * 32 + lane) * PITCH + wave * 4] = pk.q; // ds_store_b64
        }
    };

    // ---- A fragment bases in the swizzled layout ----
    int aBase[4];
    #pragma unroll
    for (int f = 0; f < 4; ++f)
        aBase[f] = (mWave * 4 + f) * (KITERS * 512) + lane * 16;

    // ---- init accumulators with the bias (fused epilogue add) ----
    v8f acc[4][4];
    #pragma unroll
    for (int f = 0; f < 4; ++f) {
        int oBase = mWave * 64 + f * 16 + hi * 8;
        float4 b0 = *(const float4*)&bias[oBase];
        float4 b1 = *(const float4*)&bias[oBase + 4];
        v8f bi;
        bi[0] = b0.x; bi[1] = b0.y; bi[2] = b0.z; bi[3] = b0.w;
        bi[4] = b1.x; bi[5] = b1.y; bi[6] = b1.z; bi[7] = b1.w;
        #pragma unroll
        for (int g = 0; g < 4; ++g) acc[f][g] = bi;
    }

    gather(0, 0);

    for (int kt = 0; kt < KITERS; ++kt) {
        __syncthreads();   // buffer (kt&1) now fully written; other buffer free
        const int cur = kt & 1;
        const int ktn = (kt + 1 < KITERS) ? kt + 1 : kt;
        gather(ktn, cur ^ 1);            // overlaps with WMMAs below

        union BFrag { uint4 q[2]; v16bf v; };
        BFrag aF[4];
        #pragma unroll
        for (int f = 0; f < 4; ++f) {
            const uint16_t* ap = wswz + aBase[f] + kt * 512;
            aF[f].q[0] = *(const uint4*)(ap);       // contiguous, fully coalesced
            aF[f].q[1] = *(const uint4*)(ap + 8);
        }
        #pragma unroll
        for (int g = 0; g < 4; ++g) {
            int n = nWave * 64 + g * 16 + ml;
            BFrag bF;
            bF.q[0] = *(const uint4*)&Bs[cur][n * PITCH + hi * 8];
            bF.q[1] = *(const uint4*)&Bs[cur][n * PITCH + 16 + hi * 8];
            #pragma unroll
            for (int f = 0; f < 4; ++f)
                acc[f][g] = __builtin_amdgcn_wmma_f32_16x16x32_bf16(
                    false, aF[f].v, false, bF.v, (short)0, acc[f][g], false, false);
        }
    }

    // ---- store: D layout — lane = column n (ml), VGPR v = row m = v + 8*hi ----
    #pragma unroll
    for (int g = 0; g < 4; ++g) {
        int n   = nBase + nWave * 64 + g * 16 + ml;
        int onn = n / HWSQ;
        int ohw = n - onn * HWSQ;
        #pragma unroll
        for (int f = 0; f < 4; ++f) {
            int oBase = mWave * 64 + f * 16 + hi * 8;
            #pragma unroll
            for (int v = 0; v < 8; ++v) {
                __builtin_nontemporal_store(acc[f][g][v],
                    &out[((size_t)onn * O_CH + oBase + v) * HWSQ + ohw]);
            }
        }
    }
}

extern "C" void kernel_launch(void* const* d_in, const int* in_sizes, int n_in,
                              void* d_out, int out_size, void* d_ws, size_t ws_size,
                              hipStream_t stream) {
    const float* input = (const float*)d_in[0];
    const int*   widx  = (const int*)d_in[1];
    const float* lut   = (const float*)d_in[2];
    const float* bias  = (const float*)d_in[3];
    float*    out  = (float*)d_out;
    uint16_t* wswz = (uint16_t*)d_ws;  // 1.13 MB swizzled weights, L2-resident

    const int    nW      = O_CH * KDIM;                   // 589824
    const size_t WBYTES  = (size_t)nW * 2;                // 1,179,648 (256B-aligned)
    const int    nPad    = 32 * C_CH * PADSQ;             // 27,557,888 bf16
    const size_t NEEDED  = WBYTES + (size_t)nPad * 2;     // ~56.3 MB

    dequant_kernel<<<(nW + 255) / 256, 256, 0, stream>>>(widx, lut, wswz, nW);

    dim3 grid(100352 / NTILE);         // 784 spatial tiles, full O per block

    if (ws_size >= NEEDED) {
        uint16_t* pin = (uint16_t*)((char*)d_ws + WBYTES);
        pad_kernel<<<(nPad + 255) / 256, 256, 0, stream>>>(input, pin, nPad);
        conv_wmma_kernel<true><<<grid, 256, 0, stream>>>(nullptr, pin, wswz, bias, out);
    } else {
        conv_wmma_kernel<false><<<grid, 256, 0, stream>>>(input, nullptr, wswz, bias, out);
    }
}
